// _ReservoirExtractor_17678085390486
// MI455X (gfx1250) — compile-verified, matrix operand order checked by reading, not run on hardware
//
#include <hip/hip_runtime.h>
#include <hip/hip_bf16.h>

// ---------------------------------------------------------------------------
// Reservoir RNN, persistent-kernel bf16 WMMA implementation for gfx1250.
//
// Sizes (fixed by the reference): V=32000 E=256 R=1024 B=64 T=2048 L=3
//
//  * Weights converted once fp32 -> bf16, pre-tiled into the WMMA B-fragment
//    per-lane layout (each 32x16 fragment = 1KB, two b128 loads per lane).
//  * Persistent grid: 32 WGs x 256 threads (8 wave32s); each wave owns one
//    16x16 C tile of the [64 x 1024] per-layer GEMM (M=64 -> 4 tiles,
//    2 N-tiles per WG). K-loop = v_wmma_f32_16x16x32_bf16, software-pipelined
//    (fragment kt+1 loads issued before the WMMA on fragment kt).
//  * Recurrent weights R0/R1/R2 N-slices (3 x 64KB) parked in LDS.
//  * Win input-weight N-slice staged into LDS each layer by the Tensor Data
//    Mover (tensor_load_to_lds), overlapped with the recurrent WMMA loop,
//    completed with s_wait_tensorcnt. 256KB LDS total (<=320KB/WGP).
//  * bf16 state ping-pong buffers (native v_cvt bf16 converts); tanh in
//    fp32; device-wide generation barrier after every layer; layernorm
//    epilogue kernel.
// ---------------------------------------------------------------------------

typedef __attribute__((ext_vector_type(16))) __bf16 v16bf;
typedef __attribute__((ext_vector_type(8)))  float  v8f;
typedef __attribute__((ext_vector_type(4)))  unsigned v4u;
typedef __attribute__((ext_vector_type(8)))  int    v8i;
typedef __attribute__((ext_vector_type(4)))  int    v4i;

#define RNN_V     32000
#define RNN_E     256
#define RNN_R     1024
#define RNN_B     64
#define RNN_T     2048
#define RNN_NWG   32

#define LDS_R_BYTES     (3 * 2 * 32 * 1024)   // 192KB: 3 layers * 2 nt * 32 frags
#define LDS_STAGE_OFF   LDS_R_BYTES
#define LDS_STAGE_BYTES (64 * 1024)           // Win slice staging (<= 64KB)
#define LDS_BYTES       (LDS_R_BYTES + LDS_STAGE_BYTES)   // 256KB

// ---- Tensor Data Mover availability / arity detection --------------------
#if defined(__has_builtin)
#  if __has_builtin(__builtin_amdgcn_tensor_load_to_lds)
#    define HAVE_TDM 1
#  endif
#endif
#if __has_include(<hip/amd_detail/amd_gfx1250_TDM.h>)
#  define TDM_SIX_ARGS 1      // therock-10.0 headers -> 6-arg builtin
#endif

union FragB {
  v16bf    v;
  unsigned u[8];
  uint4    q[2];
};

// Native bf16 converts (v_cvt_bf16_f32 / v_cvt_pk_bf16_f32), RNE.
__device__ __forceinline__ unsigned short f2bf(float f) {
  union { __bf16 h; unsigned short u; } p;
  p.h = (__bf16)f;
  return p.u;
}
__device__ __forceinline__ unsigned pack2_bf16(float lo, float hi) {
  union { __bf16 h[2]; unsigned u; } p;
  p.h[0] = (__bf16)lo;
  p.h[1] = (__bf16)hi;
  return p.u;
}

// A fragment (16x32 bf16): lane half h = lane>>4, row = lane&15,
// VGPR v holds bf16 pair (k,k+1), k = kt*32 + 2v + (v>=4 ? 8 : 0) + 8h.
__device__ __forceinline__ v16bf load_a_state(const unsigned short* __restrict__ S,
                                              int row, int kt, int lane) {
  FragB f;
  const unsigned* p = (const unsigned*)(S + (size_t)row * RNN_R);
  const int kb = kt * 16 + ((lane >> 4) << 2);
  f.u[0] = p[kb + 0];  f.u[1] = p[kb + 1];
  f.u[2] = p[kb + 2];  f.u[3] = p[kb + 3];
  f.u[4] = p[kb + 8];  f.u[5] = p[kb + 9];
  f.u[6] = p[kb + 10]; f.u[7] = p[kb + 11];
  return f.v;
}

// A fragment gathered from the fp32 embedding table (layer-0 input path).
__device__ __forceinline__ v16bf load_a_embed(const float* __restrict__ embed,
                                              int tok, int kt, int lane) {
  FragB f;
  const float* p = embed + (size_t)tok * RNN_E;
  const int half = lane >> 4;
#pragma unroll
  for (int v = 0; v < 8; ++v) {
    int k = kt * 32 + 2 * v + (v >= 4 ? 8 : 0) + 8 * half;
    f.u[v] = pack2_bf16(p[k], p[k + 1]);   // v_cvt_pk_bf16_f32
  }
  return f.v;
}

// B fragment from pre-tiled storage: fragment = 1KB = 32 lanes x 32B.
__device__ __forceinline__ v16bf load_b_frag(const uint4* __restrict__ base,
                                             int frag, int lane) {
  FragB f;
  const uint4* p = base + ((size_t)frag << 6) + (lane << 1);
  f.q[0] = p[0];
  f.q[1] = p[1];
  return f.v;
}

#define WMMA_BF16(a, b, c) \
  __builtin_amdgcn_wmma_f32_16x16x32_bf16(false, (a), false, (b), (short)0, (c), false, false)

// ---------------------------------------------------------------------------
// TDM: 1D DMA of `bytes` (multiple of 8) from global `gsrc` into LDS offset
// `lds_off`. D# built per CDNA5 ISA ch.8: group0 {count, lds_addr,
// global_addr, type=2}; group1 {data_size=8B, tensor_dim0, tile_dim0,
// stride}; groups 2/3 zero (<=2D tensor).
// ---------------------------------------------------------------------------
#if HAVE_TDM
__device__ __forceinline__ void tdm_load_1d(unsigned lds_off, const void* gsrc,
                                            unsigned bytes) {
  const unsigned n = bytes >> 3;                         // 8-byte elements
  const unsigned long long ga = (unsigned long long)gsrc;
  v4u g0;
  g0.x = 1u;                                             // count=1, user mode
  g0.y = lds_off;                                        // lds_addr (bytes)
  g0.z = (unsigned)(ga & 0xFFFFFFFFu);                   // global_addr[31:0]
  g0.w = (unsigned)((ga >> 32) & 0x01FFFFFFu) | (2u << 30);  // addr[56:32]|type=2
  v8i g1;
  g1[0] = (int)(3u << 16);                 // wg_mask=0, data_size=3 (8B)
  g1[1] = (int)((n & 0xFFFFu) << 16);      // tensor_dim0[15:0]
  g1[2] = (int)(((n >> 16) & 0xFFFFu) | (1u << 16));  // dim0[31:16], dim1=1 lo
  g1[3] = (int)((n & 0xFFFFu) << 16);      // tensor_dim1 hi=0, tile_dim0=n
  g1[4] = 0;                               // tile_dim1=0, tile_dim2=0
  g1[5] = (int)n;                          // tensor_dim0_stride[31:0]
  g1[6] = 0;
  g1[7] = 0;
  v4i z4 = {0, 0, 0, 0};
#if TDM_SIX_ARGS
  v8i z8 = {0, 0, 0, 0, 0, 0, 0, 0};
  __builtin_amdgcn_tensor_load_to_lds(g0, g1, z4, z4, z8, 0);
#else
  __builtin_amdgcn_tensor_load_to_lds(g0, g1, z4, z4, 0);
#endif
}
#endif

// ---------------------------------------------------------------------------
// Device-wide barrier: generation counter in global scratch.
// ---------------------------------------------------------------------------
__device__ __forceinline__ void grid_barrier(int* cnt, int* gen) {
  __syncthreads();
  __threadfence();
  if (threadIdx.x == 0) {
    int g = *(volatile int*)gen;
    if (atomicAdd(cnt, 1) == RNN_NWG - 1) {
      *(volatile int*)cnt = 0;
      __threadfence();
      atomicAdd(gen, 1);
    } else {
      while (*(volatile int*)gen == g) { __builtin_amdgcn_s_sleep(2); }
    }
  }
  __threadfence();
  __syncthreads();
}

// ---------------------------------------------------------------------------
// Weight re-tiling: row-major fp32 [K x N] -> tiled bf16 fragments.
// ---------------------------------------------------------------------------
__global__ void tile_weights(const float* __restrict__ src,
                             unsigned short* __restrict__ dst,
                             int K, int N) {
  int tid = blockIdx.x * blockDim.x + threadIdx.x;
  int total = (K * N) >> 1;
  if (tid >= total) return;
  int v    = tid & 7;
  int lane = (tid >> 3) & 31;
  int frag = tid >> 8;
  int ktn  = K >> 5;
  int kt   = frag % ktn;
  int nt   = frag / ktn;
  int half = lane >> 4;
  int n = nt * 16 + (lane & 15);
  int k = kt * 32 + 2 * v + (v >= 4 ? 8 : 0) + 8 * half;
  ((unsigned*)dst)[(size_t)frag * 256 + lane * 8 + v] =
      pack2_bf16(src[(size_t)k * N + n], src[(size_t)(k + 1) * N + n]);
}

__global__ void zero_init(unsigned short* s0, unsigned short* s1, int n, int* sync) {
  int i = blockIdx.x * blockDim.x + threadIdx.x;
  if (i < n) { s0[i] = 0; s1[i] = 0; }
  if (i < 2) sync[i] = 0;
}

// ---------------------------------------------------------------------------
// Persistent RNN kernel.
// ---------------------------------------------------------------------------
__global__ void __launch_bounds__(256, 1)
rnn_kernel(const int* __restrict__ x, const float* __restrict__ embed,
           const unsigned short* __restrict__ w_win0,   // tiled, K=256
           const unsigned short* __restrict__ w_win1,   // tiled, K=1024
           const unsigned short* __restrict__ w_win2,
           const unsigned short* __restrict__ w_r0,
           const unsigned short* __restrict__ w_r1,
           const unsigned short* __restrict__ w_r2,
           unsigned short* __restrict__ s0buf,
           unsigned short* __restrict__ s1buf,
           float* __restrict__ hbuf, int* __restrict__ sync) {
  extern __shared__ unsigned char smem_raw[];
  const int tid  = threadIdx.x;
  const int lane = tid & 31;
  const int wave = tid >> 5;
  const int wg   = blockIdx.x;
  const int mt   = wave >> 1;       // M tile 0..3
  const int ntl  = wave & 1;        // local N tile 0..1
  const int ntg  = wg * 2 + ntl;    // global N tile 0..63

  // ---- Park the recurrent-weight N-slices in LDS (3 x 2 x 32 frags). ----
  {
    const unsigned short* srcs[3] = { w_r0, w_r1, w_r2 };
    for (int l = 0; l < 3; ++l) {
      const uint4* src = (const uint4*)srcs[l] + (size_t)(wg * 2) * 32 * 64;
      uint4* dst = (uint4*)smem_raw + (size_t)l * 2 * 32 * 64;
      for (int i = tid; i < 2 * 32 * 64; i += 256) dst[i] = src[i];
    }
  }
  __syncthreads();

  const uint4* stage = (const uint4*)(smem_raw + LDS_STAGE_OFF);
  const int arow = mt * 16 + (lane & 15);     // batch row this lane feeds to A
  const int half = lane >> 4;
  const int ncol = ntg * 16 + (lane & 15);    // output column this lane owns

  for (int t = 0; t < RNN_T; ++t) {
    unsigned short* cur  = (t & 1) ? s1buf : s0buf;
    unsigned short* prev = (t & 1) ? s0buf : s1buf;
    const int tok = x[arow * RNN_T + t];

    for (int layer = 0; layer < 3; ++layer) {
#if HAVE_TDM
      // Kick the TDM: stage this WG's Win[layer] N-slice into LDS while the
      // recurrent WMMA loop runs. Layer0 slice = 16KB, layers 1/2 = 64KB.
      if (wave == 0) {
        if (layer == 0)
          tdm_load_1d(LDS_STAGE_OFF, (const char*)w_win0 + (size_t)wg * 16 * 1024,
                      16 * 1024);
        else
          tdm_load_1d(LDS_STAGE_OFF,
                      (const char*)((layer == 1) ? w_win1 : w_win2) +
                          (size_t)wg * 64 * 1024,
                      64 * 1024);
      }
#endif
      v8f acc = {};

      // --- recurrent path: states_prev[layer] @ R[layer], K = 1024 ---
      // Software-pipelined: issue fragment kt+1 loads before WMMA on kt.
      {
        const unsigned short* Sprev = prev + (size_t)layer * RNN_B * RNN_R;
        const uint4* BR =
            (const uint4*)smem_raw + (size_t)(layer * 2 + ntl) * 32 * 64;
        v16bf a = load_a_state(Sprev, arow, 0, lane);
        v16bf b = load_b_frag(BR, 0, lane);
#pragma unroll 4
        for (int kt = 0; kt < 31; ++kt) {
          v16bf an = load_a_state(Sprev, arow, kt + 1, lane);
          v16bf bn = load_b_frag(BR, kt + 1, lane);    // ds_load_b128
          acc = WMMA_BF16(a, b, acc);
          a = an;
          b = bn;
        }
        acc = WMMA_BF16(a, b, acc);
      }

#if HAVE_TDM
      // Complete the staged Win slice before the input loop.
      if (wave == 0) __builtin_amdgcn_s_wait_tensorcnt(0);
      __syncthreads();
#endif

      // --- input path ---
      if (layer == 0) {
        // xt @ Win0, K = 256, A gathered from the fp32 embedding table.
#if HAVE_TDM
        const uint4* BW = stage + (size_t)ntl * 8 * 64;
#else
        const uint4* BW = (const uint4*)w_win0 + (size_t)ntg * 8 * 64;
        __builtin_prefetch(BW, 0, 1);
#endif
        v16bf a = load_a_embed(embed, tok, 0, lane);
        v16bf b = load_b_frag(BW, 0, lane);
#pragma unroll
        for (int kt = 0; kt < 7; ++kt) {
          v16bf an = load_a_embed(embed, tok, kt + 1, lane);
          v16bf bn = load_b_frag(BW, kt + 1, lane);
          acc = WMMA_BF16(a, b, acc);
          a = an;
          b = bn;
        }
        acc = WMMA_BF16(a, b, acc);
      } else {
        // fresh lower-layer state @ Win[layer], K = 1024
        const unsigned short* Scur = cur + (size_t)(layer - 1) * RNN_B * RNN_R;
#if HAVE_TDM
        const uint4* BW = stage + (size_t)ntl * 32 * 64;
#else
        const uint4* BW = (const uint4*)((layer == 1) ? w_win1 : w_win2)
                          + (size_t)ntg * 32 * 64;
        __builtin_prefetch(BW, 0, 1);
#endif
        v16bf a = load_a_state(Scur, arow, 0, lane);
        v16bf b = load_b_frag(BW, 0, lane);
#pragma unroll 4
        for (int kt = 0; kt < 31; ++kt) {
          v16bf an = load_a_state(Scur, arow, kt + 1, lane);
          v16bf bn = load_b_frag(BW, kt + 1, lane);
          acc = WMMA_BF16(a, b, acc);
          a = an;
          b = bn;
        }
        acc = WMMA_BF16(a, b, acc);
      }

      // --- tanh + store new state (C/D layout: VGPR r -> row r + 8*half) ---
      unsigned short* Sout = cur + (size_t)layer * RNN_B * RNN_R;
      const bool last = (layer == 2) && (t == RNN_T - 1);
#pragma unroll
      for (int r = 0; r < 8; ++r) {
        int grow = mt * 16 + r + 8 * half;
        float val = tanhf(acc[r]);
        Sout[(size_t)grow * RNN_R + ncol] = f2bf(val);    // v_cvt_bf16_f32
        if (last) hbuf[(size_t)grow * RNN_R + ncol] = val;
      }

      grid_barrier(sync, sync + 1);
    }
  }
}

// ---------------------------------------------------------------------------
// LayerNorm epilogue over the final layer-2 state: [64 x 1024].
// ---------------------------------------------------------------------------
__global__ void ln_kernel(const float* __restrict__ h,
                          const float* __restrict__ lw,
                          const float* __restrict__ lb,
                          float* __restrict__ out) {
  const int row = blockIdx.x;
  const int tid = threadIdx.x;
  __shared__ float red0[8], red1[8];
  float s = 0.f, s2 = 0.f;
  for (int i = tid; i < RNN_R; i += 256) {
    float v = h[(size_t)row * RNN_R + i];
    s += v; s2 += v * v;
  }
#pragma unroll
  for (int off = 16; off > 0; off >>= 1) {
    s  += __shfl_down(s, off);
    s2 += __shfl_down(s2, off);
  }
  if ((tid & 31) == 0) { red0[tid >> 5] = s; red1[tid >> 5] = s2; }
  __syncthreads();
  if (tid == 0) {
    float a = 0.f, b = 0.f;
    for (int i = 0; i < 8; ++i) { a += red0[i]; b += red1[i]; }
    red0[0] = a; red1[0] = b;
  }
  __syncthreads();
  float mu   = red0[0] * (1.0f / RNN_R);
  float var  = red1[0] * (1.0f / RNN_R) - mu * mu;
  float rstd = rsqrtf(var + 1e-5f);
  for (int i = tid; i < RNN_R; i += 256) {
    float v = h[(size_t)row * RNN_R + i];
    out[(size_t)row * RNN_R + i] = (v - mu) * rstd * lw[i] + lb[i];
  }
}

// ---------------------------------------------------------------------------
// Host launcher.
// ---------------------------------------------------------------------------
extern "C" void kernel_launch(void* const* d_in, const int* in_sizes, int n_in,
                              void* d_out, int out_size, void* d_ws, size_t ws_size,
                              hipStream_t stream) {
  const int*   x      = (const int*)  d_in[0];
  const float* embed  = (const float*)d_in[1];
  const float* Win0   = (const float*)d_in[2];
  const float* Win1   = (const float*)d_in[3];
  const float* Win2   = (const float*)d_in[4];
  const float* R0     = (const float*)d_in[5];
  const float* R1     = (const float*)d_in[6];
  const float* R2     = (const float*)d_in[7];
  const float* ln_w   = (const float*)d_in[8];
  const float* ln_b   = (const float*)d_in[9];
  float* out = (float*)d_out;

  auto align = [](size_t v) { return (v + 255) & ~size_t(255); };
  char* ws = (char*)d_ws;
  size_t off = 0;
  unsigned short* w0 = (unsigned short*)(ws + off); off = align(off + (size_t)RNN_E * RNN_R * 2);
  unsigned short* w1 = (unsigned short*)(ws + off); off = align(off + (size_t)RNN_R * RNN_R * 2);
  unsigned short* w2 = (unsigned short*)(ws + off); off = align(off + (size_t)RNN_R * RNN_R * 2);
  unsigned short* r0 = (unsigned short*)(ws + off); off = align(off + (size_t)RNN_R * RNN_R * 2);
  unsigned short* r1 = (unsigned short*)(ws + off); off = align(off + (size_t)RNN_R * RNN_R * 2);
  unsigned short* r2 = (unsigned short*)(ws + off); off = align(off + (size_t)RNN_R * RNN_R * 2);
  unsigned short* s0 = (unsigned short*)(ws + off); off = align(off + (size_t)3 * RNN_B * RNN_R * 2);
  unsigned short* s1 = (unsigned short*)(ws + off); off = align(off + (size_t)3 * RNN_B * RNN_R * 2);
  float*          hb = (float*)(ws + off);          off = align(off + (size_t)RNN_B * RNN_R * 4);
  int*          sync = (int*)(ws + off);            off = align(off + 64);
  (void)ws_size; (void)n_in; (void)in_sizes; (void)out_size;

  {
    int n, g;
    n = (RNN_E * RNN_R) / 2; g = (n + 255) / 256;
    tile_weights<<<g, 256, 0, stream>>>(Win0, w0, RNN_E, RNN_R);
    n = (RNN_R * RNN_R) / 2; g = (n + 255) / 256;
    tile_weights<<<g, 256, 0, stream>>>(Win1, w1, RNN_R, RNN_R);
    tile_weights<<<g, 256, 0, stream>>>(Win2, w2, RNN_R, RNN_R);
    tile_weights<<<g, 256, 0, stream>>>(R0,   r0, RNN_R, RNN_R);
    tile_weights<<<g, 256, 0, stream>>>(R1,   r1, RNN_R, RNN_R);
    tile_weights<<<g, 256, 0, stream>>>(R2,   r2, RNN_R, RNN_R);
  }

  {
    int n = 3 * RNN_B * RNN_R;
    zero_init<<<(n + 255) / 256, 256, 0, stream>>>(s0, s1, n, sync);
  }

  hipFuncSetAttribute((const void*)rnn_kernel,
                      hipFuncAttributeMaxDynamicSharedMemorySize, LDS_BYTES);
  rnn_kernel<<<RNN_NWG, 256, LDS_BYTES, stream>>>(x, embed, w0, w1, w2,
                                                  r0, r1, r2, s0, s1, hb, sync);

  ln_kernel<<<RNN_B, 256, 0, stream>>>(hb, ln_w, ln_b, out);
}
